// AiLUT_82755429859565
// MI455X (gfx1250) — matrix operand order repaired and checked by reading, not compiled, not run on hardware
//
#include <hip/hip_runtime.h>
#include <hip/hip_bf16.h>

typedef __attribute__((ext_vector_type(16))) _Float16 v16h;
typedef __attribute__((ext_vector_type(8)))  float    v8f;

#define LUT_V 33
#define NB 4
#define IMG_H 1080
#define IMG_W 1920

// ---------------------------------------------------------------------------
// 1) Bilinear resize (antialias=false): out (4,3,256,256)
// ---------------------------------------------------------------------------
__global__ void resize_kernel(const float* __restrict__ imgs, float* __restrict__ out) {
    int idx = blockIdx.x * blockDim.x + threadIdx.x;
    if (idx >= NB * 3 * 256 * 256) return;
    int ox = idx & 255;
    int t  = idx >> 8;
    int oy = t & 255;
    t >>= 8;                      // t = b*3 + c
    const float* src = imgs + (long)t * IMG_H * IMG_W;
    float sy = (oy + 0.5f) * ((float)IMG_H / 256.0f) - 0.5f;
    float sx = (ox + 0.5f) * ((float)IMG_W / 256.0f) - 0.5f;
    int y0 = (int)floorf(sy); float fy = sy - (float)y0;
    int x0 = (int)floorf(sx); float fx = sx - (float)x0;
    int y1 = min(y0 + 1, IMG_H - 1); y0 = max(y0, 0);
    int x1 = min(x0 + 1, IMG_W - 1); x0 = max(x0, 0);
    float p00 = src[(long)y0 * IMG_W + x0];
    float p01 = src[(long)y0 * IMG_W + x1];
    float p10 = src[(long)y1 * IMG_W + x0];
    float p11 = src[(long)y1 * IMG_W + x1];
    out[idx] = (1.f - fy) * ((1.f - fx) * p00 + fx * p01)
             +        fy  * ((1.f - fx) * p10 + fx * p11);
}

// ---------------------------------------------------------------------------
// 2) Direct f32 conv (stride 2, pad 1, 3x3) + bias + leaky-relu (conv1, conv2)
// ---------------------------------------------------------------------------
__global__ void conv_direct_kernel(const float* __restrict__ x, const float* __restrict__ w,
                                   const float* __restrict__ bias, float* __restrict__ y,
                                   int Cin, int Cout, int Hin, int Win, int Hout, int Wout) {
    long idx   = (long)blockIdx.x * blockDim.x + threadIdx.x;
    long total = (long)NB * Cout * Hout * Wout;
    if (idx >= total) return;
    int ox = (int)(idx % Wout); long t = idx / Wout;
    int oy = (int)(t % Hout);   t /= Hout;
    int co = (int)(t % Cout);
    int b  = (int)(t / Cout);
    float acc = bias[co];
    const float* xb = x + (long)b * Cin * Hin * Win;
    for (int ci = 0; ci < Cin; ci++) {
        for (int ky = 0; ky < 3; ky++) {
            int iy = oy * 2 - 1 + ky;
            if (iy < 0 || iy >= Hin) continue;
            for (int kx = 0; kx < 3; kx++) {
                int ix = ox * 2 - 1 + kx;
                if (ix < 0 || ix >= Win) continue;
                acc += xb[((long)ci * Hin + iy) * Win + ix] *
                       w[((co * Cin + ci) * 3 + ky) * 3 + kx];
            }
        }
    }
    y[idx] = (acc > 0.f) ? acc : 0.2f * acc;
}

// ---------------------------------------------------------------------------
// 3) WMMA implicit-GEMM conv (conv3/4/5): one wave computes a 16x16 (MxN)
//    output tile, K = Cin*9 (multiple of 32). f16 in, f32 accumulate.
//    Dual accumulators over interleaved K-chunks to hide WMMA RAW latency.
// ---------------------------------------------------------------------------
__device__ __forceinline__ void build_frags(
        const float* __restrict__ xb, const float* __restrict__ wrow,
        int k0, int half, int oy, int ox, int Hin, int Win,
        v16h& afrag, v16h& bfrag) {
#pragma unroll
    for (int e = 0; e < 16; e++) {
        // A-matrix 16x32 f16 layout: lanes 0-15 hold K={0..7,16..23},
        // lanes 16-31 hold K={8..15,24..31}
        int ka = k0 + e + 8 * half + ((e >= 8) ? 8 : 0);
        int ci = ka / 9;
        int r  = ka - ci * 9;
        int ky = r / 3;
        int kx = r - ky * 3;
        int iy = oy * 2 - 1 + ky;
        int ix = ox * 2 - 1 + kx;
        bool inb = (iy >= 0) & (iy < Hin) & (ix >= 0) & (ix < Win);
        float av = inb ? xb[((long)ci * Hin + iy) * Win + ix] : 0.f;
        afrag[e] = (_Float16)av;
        // B-matrix 32x16 f16 layout: same K order as A (lanes 0-15 K={0..7,16..23}, lanes 16-31 K={8..15,24..31})
        int kb = k0 + e + 8 * half + ((e >= 8) ? 8 : 0);
        bfrag[e] = (_Float16)wrow[kb];
    }
}

__global__ void __launch_bounds__(32)
conv_wmma_kernel(const float* __restrict__ x, const float* __restrict__ w,
                 const float* __restrict__ bias, float* __restrict__ y,
                 int Cin, int Cout, int Hin, int Win, int Hout, int Wout) {
    const int lane  = threadIdx.x;
    const int half  = lane >> 4;
    const int lw    = lane & 15;
    const int b     = blockIdx.z;
    const int mbase = blockIdx.x * 16;
    const int nbase = blockIdx.y * 16;
    const int K     = Cin * 9;

    const int m  = mbase + lw;           // A row held by this lane
    const int oy = m / Wout;
    const int ox = m - oy * Wout;
    const int co = nbase + lw;           // B column (output channel)

    const float* xb   = x + (long)b * Cin * Hin * Win;
    const float* wrow = w + (long)co * K;  // OIHW flat: w[co][k], k=(ci*3+ky)*3+kx

    v8f acc0 = {}, acc1 = {};
    int k0 = 0;
    for (; k0 + 64 <= K; k0 += 64) {
        v16h a0, b0, a1, b1;
        build_frags(xb, wrow, k0,      half, oy, ox, Hin, Win, a0, b0);
        build_frags(xb, wrow, k0 + 32, half, oy, ox, Hin, Win, a1, b1);
        acc0 = __builtin_amdgcn_wmma_f32_16x16x32_f16(
                   false, a0, false, b0, (short)0, acc0, false, false);
        acc1 = __builtin_amdgcn_wmma_f32_16x16x32_f16(
                   false, a1, false, b1, (short)0, acc1, false, false);
    }
    for (; k0 < K; k0 += 32) {           // remainder (K % 64 == 32, e.g. conv3)
        v16h a0, b0;
        build_frags(xb, wrow, k0, half, oy, ox, Hin, Win, a0, b0);
        acc0 = __builtin_amdgcn_wmma_f32_16x16x32_f16(
                   false, a0, false, b0, (short)0, acc0, false, false);
    }

    const float bc = bias[co];
#pragma unroll
    for (int i = 0; i < 8; i++) {
        // D layout: element i -> (M = i + 8*half, N = lw)
        int mm  = mbase + 8 * half + i;
        int oy2 = mm / Wout;
        int ox2 = mm - oy2 * Wout;
        float v = acc0[i] + acc1[i] + bc;
        v = (v > 0.f) ? v : 0.2f * v;
        y[(((long)b * Cout + co) * Hout + oy2) * Wout + ox2] = v;
    }
}

// ---------------------------------------------------------------------------
// 4) Instance norm (per b,c over HxW), in-place. One block per (b,c).
// ---------------------------------------------------------------------------
__global__ void instnorm_kernel(float* __restrict__ y, const float* __restrict__ g,
                                const float* __restrict__ be, int C, int HW) {
    int bc = blockIdx.x;
    int c  = bc % C;
    float* p = y + (long)bc * HW;
    __shared__ float s1[256], s2[256];
    float sum = 0.f, sq = 0.f;
    for (int i = threadIdx.x; i < HW; i += blockDim.x) {
        float v = p[i];
        sum += v;
        sq  += v * v;
    }
    s1[threadIdx.x] = sum;
    s2[threadIdx.x] = sq;
    __syncthreads();
    for (int s = 128; s > 0; s >>= 1) {
        if ((int)threadIdx.x < s) {
            s1[threadIdx.x] += s1[threadIdx.x + s];
            s2[threadIdx.x] += s2[threadIdx.x + s];
        }
        __syncthreads();
    }
    float mean = s1[0] / (float)HW;
    float var  = s2[0] / (float)HW - mean * mean;
    float sc = rsqrtf(var + 1e-5f) * g[c];
    float sh = be[c] - mean * sc;
    for (int i = threadIdx.x; i < HW; i += blockDim.x)
        p[i] = p[i] * sc + sh;
}

// ---------------------------------------------------------------------------
// 5) 4x4 average pool: (4,128,8,8) -> feats (4,512), f = c*4 + ph*2 + pw
// ---------------------------------------------------------------------------
__global__ void pool_kernel(const float* __restrict__ a5, float* __restrict__ feats) {
    int idx = blockIdx.x * blockDim.x + threadIdx.x;
    if (idx >= NB * 512) return;
    int pw = idx & 1;
    int ph = (idx >> 1) & 1;
    int c  = (idx >> 2) & 127;
    int b  = idx >> 9;
    const float* p = a5 + (((long)b * 128 + c) * 8 + ph * 4) * 8 + pw * 4;
    float s = 0.f;
    for (int i = 0; i < 4; i++)
        for (int j = 0; j < 4; j++)
            s += p[i * 8 + j];
    feats[idx] = s * (1.f / 16.f);
}

// ---------------------------------------------------------------------------
// 6) FC: out[b][o] = feats[b] . W[o] + bias[o]   (ada: O=96, wg: O=3)
// ---------------------------------------------------------------------------
__global__ void fc_kernel(const float* __restrict__ feats, const float* __restrict__ W,
                          const float* __restrict__ bias, float* __restrict__ out, int O) {
    int idx = blockIdx.x * blockDim.x + threadIdx.x;
    if (idx >= NB * O) return;
    int o = idx % O;
    int b = idx / O;
    const float* f = feats + (long)b * 512;
    const float* w = W + (long)o * 512;
    float acc = bias[o];
    for (int k = 0; k < 512; k++) acc += f[k] * w[k];
    out[idx] = acc;
}

// ---------------------------------------------------------------------------
// 7) softmax(32) + cumsum + leading-zero pad -> vertices (4,3,33)
// ---------------------------------------------------------------------------
__global__ void vertices_kernel(const float* __restrict__ ints, float* __restrict__ vertices) {
    int idx = threadIdx.x;        // one thread per (b, channel) row
    if (idx >= NB * 3) return;
    const float* row = ints + idx * (LUT_V - 1);
    float m = row[0];
    for (int k = 1; k < LUT_V - 1; k++) m = fmaxf(m, row[k]);
    float e[LUT_V - 1];
    float s = 0.f;
    for (int k = 0; k < LUT_V - 1; k++) { e[k] = __expf(row[k] - m); s += e[k]; }
    float inv = 1.f / s;
    float* v = vertices + idx * LUT_V;
    v[0] = 0.f;
    float cum = 0.f;
    for (int k = 0; k < LUT_V - 1; k++) { cum += e[k] * inv; v[k + 1] = cum; }
}

// ---------------------------------------------------------------------------
// 8) LUT generation: luts[b][j] = sum_t wts[b][t] * basis_w[j][t]
// ---------------------------------------------------------------------------
__global__ void lut_kernel(const float* __restrict__ wts, const float* __restrict__ basis_w,
                           float* __restrict__ luts) {
    const long N = 3L * LUT_V * LUT_V * LUT_V;
    long idx = (long)blockIdx.x * blockDim.x + threadIdx.x;
    if (idx >= NB * N) return;
    int  b = (int)(idx / N);
    long j = idx % N;
    const float* bw = basis_w + j * 3;
    const float* wb = wts + b * 3;
    luts[idx] = wb[0] * bw[0] + wb[1] * bw[1] + wb[2] * bw[2];
}

// ---------------------------------------------------------------------------
// 9) Per-pixel apply: 4 pixels/thread via float4, branchless binary
//    searchsorted on LDS vertices, precomputed segment reciprocals,
//    7-lerp trilinear with corner indices shared across output channels.
// ---------------------------------------------------------------------------
__device__ __forceinline__ float lerpf(float a, float b, float f) {
    return fmaf(f, b - a, a);
}

__global__ void apply_kernel(const float4* __restrict__ imgs,
                             const float* __restrict__ vertices,
                             const float* __restrict__ luts,
                             float4* __restrict__ out, long P4) {
    const int b = blockIdx.y;
    __shared__ float sv[3 * LUT_V];
    __shared__ float siw[3 * (LUT_V - 1)];
    for (int t = threadIdx.x; t < 3 * LUT_V; t += blockDim.x)
        sv[t] = vertices[(long)b * 3 * LUT_V + t];
    __syncthreads();
    for (int t = threadIdx.x; t < 3 * (LUT_V - 1); t += blockDim.x) {
        int c = t / (LUT_V - 1);
        int k = t - c * (LUT_V - 1);
        siw[t] = 1.f / (sv[c * LUT_V + k + 1] - sv[c * LUT_V + k] + 1e-10f);
    }
    __syncthreads();

    long p4 = (long)blockIdx.x * blockDim.x + threadIdx.x;
    if (p4 >= P4) return;

    float4 in4[3], res[3];
#pragma unroll
    for (int c = 0; c < 3; c++)
        in4[c] = imgs[((long)(b * 3 + c)) * P4 + p4];

    const int V1 = LUT_V, V2 = LUT_V * LUT_V, V3 = LUT_V * LUT_V * LUT_V;

#pragma unroll
    for (int u = 0; u < 4; u++) {
        int   i0[3];
        float fr[3];
#pragma unroll
        for (int c = 0; c < 3; c++) {
            float q = ((const float*)&in4[c])[u];
            q = fminf(fmaxf(q, 0.f), 1.f);
            const float* v = sv + c * LUT_V;
            // branchless upper_bound over 33 sorted entries: lo = last idx with v[lo] <= q
            int lo = 0;
#pragma unroll
            for (int st = 32; st >= 1; st >>= 1) {
                int t2 = lo + st;
                float vt = v[min(t2, LUT_V - 1)];
                bool ok = (t2 < LUT_V) && (vt <= q);
                lo = ok ? t2 : lo;
            }
            int i = min(lo, LUT_V - 2);           // i0 = clip(count,1,32) - 1
            i0[c] = i;
            fr[c] = (q - v[i]) * siw[c * (LUT_V - 1) + i];
        }

        // corner indices shared across the 3 output channels
        int base = (i0[2] * V1 + i0[1]) * V1 + i0[0];

#pragma unroll
        for (int c = 0; c < 3; c++) {
            const float* L = luts + ((long)(b * 3 + c)) * V3 + base;
            float c000 = L[0],       c001 = L[1];
            float c010 = L[V1],      c011 = L[V1 + 1];
            float c100 = L[V2],      c101 = L[V2 + 1];
            float c110 = L[V2 + V1], c111 = L[V2 + V1 + 1];
            float r00 = lerpf(c000, c001, fr[0]);
            float r01 = lerpf(c010, c011, fr[0]);
            float r10 = lerpf(c100, c101, fr[0]);
            float r11 = lerpf(c110, c111, fr[0]);
            float g0  = lerpf(r00, r01, fr[1]);
            float g1  = lerpf(r10, r11, fr[1]);
            ((float*)&res[c])[u] = lerpf(g0, g1, fr[2]);
        }
    }

#pragma unroll
    for (int c = 0; c < 3; c++)
        out[((long)(b * 3 + c)) * P4 + p4] = res[c];
}

// ---------------------------------------------------------------------------
// Host launcher
// ---------------------------------------------------------------------------
extern "C" void kernel_launch(void* const* d_in, const int* in_sizes, int n_in,
                              void* d_out, int out_size, void* d_ws, size_t ws_size,
                              hipStream_t stream) {
    const float* imgs    = (const float*)d_in[0];
    const float* c1_w    = (const float*)d_in[1];
    const float* c1_b    = (const float*)d_in[2];
    const float* g1      = (const float*)d_in[3];
    const float* b1      = (const float*)d_in[4];
    const float* c2_w    = (const float*)d_in[5];
    const float* c2_b    = (const float*)d_in[6];
    const float* g2      = (const float*)d_in[7];
    const float* b2      = (const float*)d_in[8];
    const float* c3_w    = (const float*)d_in[9];
    const float* c3_b    = (const float*)d_in[10];
    const float* g3      = (const float*)d_in[11];
    const float* b3      = (const float*)d_in[12];
    const float* c4_w    = (const float*)d_in[13];
    const float* c4_b    = (const float*)d_in[14];
    const float* g4      = (const float*)d_in[15];
    const float* b4      = (const float*)d_in[16];
    const float* c5_w    = (const float*)d_in[17];
    const float* c5_b    = (const float*)d_in[18];
    const float* ada_w   = (const float*)d_in[19];
    const float* ada_b   = (const float*)d_in[20];
    const float* wg_w    = (const float*)d_in[21];
    const float* wg_b    = (const float*)d_in[22];
    const float* basis_w = (const float*)d_in[23];
    float* out = (float*)d_out;

    // workspace partition (floats)
    float* ws = (float*)d_ws;
    const long SZ_X0 = 4L * 3 * 256 * 256;
    const long SZ_A1 = 4L * 16 * 128 * 128;
    const long SZ_A2 = 4L * 32 * 64 * 64;
    const long SZ_A3 = 4L * 64 * 32 * 32;
    const long SZ_A4 = 4L * 128 * 16 * 16;
    const long SZ_A5 = 4L * 128 * 8 * 8;
    const long SZ_FE = 4L * 512;
    const long SZ_IN = 4L * 96;
    const long SZ_VT = 4L * 3 * LUT_V;
    const long SZ_WT = 4L * 3;
    float* x0   = ws;            ws += SZ_X0;
    float* a1   = ws;            ws += SZ_A1;
    float* a2   = ws;            ws += SZ_A2;
    float* a3   = ws;            ws += SZ_A3;
    float* a4   = ws;            ws += SZ_A4;
    float* a5   = ws;            ws += SZ_A5;
    float* fe   = ws;            ws += SZ_FE;
    float* ints = ws;            ws += SZ_IN;
    float* vtx  = ws;            ws += SZ_VT;
    float* wts  = ws;            ws += SZ_WT;
    float* luts = ws;            // 4*3*33^3 = 431244 floats

    const long P = (long)IMG_H * IMG_W;

    // 1) resize
    resize_kernel<<<(NB * 3 * 256 * 256 + 255) / 256, 256, 0, stream>>>(imgs, x0);

    // 2) conv1 (3->16, 256->128) direct + norm
    conv_direct_kernel<<<(int)((4L * 16 * 128 * 128 + 255) / 256), 256, 0, stream>>>(
        x0, c1_w, c1_b, a1, 3, 16, 256, 256, 128, 128);
    instnorm_kernel<<<4 * 16, 256, 0, stream>>>(a1, g1, b1, 16, 128 * 128);

    // 3) conv2 (16->32, 128->64) direct + norm
    conv_direct_kernel<<<(int)((4L * 32 * 64 * 64 + 255) / 256), 256, 0, stream>>>(
        a1, c2_w, c2_b, a2, 16, 32, 128, 128, 64, 64);
    instnorm_kernel<<<4 * 32, 256, 0, stream>>>(a2, g2, b2, 32, 64 * 64);

    // 4) conv3 (32->64, 64->32) WMMA + norm   M=1024,N=64,K=288
    {
        dim3 g(64, 4, 4);
        conv_wmma_kernel<<<g, 32, 0, stream>>>(a2, c3_w, c3_b, a3, 32, 64, 64, 64, 32, 32);
    }
    instnorm_kernel<<<4 * 64, 256, 0, stream>>>(a3, g3, b3, 64, 32 * 32);

    // 5) conv4 (64->128, 32->16) WMMA + norm  M=256,N=128,K=576
    {
        dim3 g(16, 8, 4);
        conv_wmma_kernel<<<g, 32, 0, stream>>>(a3, c4_w, c4_b, a4, 64, 128, 32, 32, 16, 16);
    }
    instnorm_kernel<<<4 * 128, 256, 0, stream>>>(a4, g4, b4, 128, 16 * 16);

    // 6) conv5 (128->128, 16->8) WMMA, no norm  M=64,N=128,K=1152
    {
        dim3 g(4, 8, 4);
        conv_wmma_kernel<<<g, 32, 0, stream>>>(a4, c5_w, c5_b, a5, 128, 128, 16, 16, 8, 8);
    }

    // 7) pool -> feats
    pool_kernel<<<(NB * 512 + 255) / 256, 256, 0, stream>>>(a5, fe);

    // 8) heads
    fc_kernel<<<(NB * 96 + 255) / 256, 256, 0, stream>>>(fe, ada_w, ada_b, ints, 96);
    fc_kernel<<<1, 256, 0, stream>>>(fe, wg_w, wg_b, wts, 3);
    vertices_kernel<<<1, 32, 0, stream>>>(ints, vtx);

    // 9) LUT generation
    {
        long n = 4L * 3 * LUT_V * LUT_V * LUT_V;
        lut_kernel<<<(int)((n + 255) / 256), 256, 0, stream>>>(wts, basis_w, luts);
    }

    // 10) per-pixel apply (memory-bound: ~200 MB traffic -> ~8.6 us floor)
    {
        const long P4 = P / 4;
        dim3 g((unsigned)((P4 + 255) / 256), NB);
        apply_kernel<<<g, 256, 0, stream>>>((const float4*)imgs, vtx, luts,
                                            (float4*)out, P4);
    }
}
